// CLIPAttention_28776280883341
// MI455X (gfx1250) — matrix operand order repaired and checked by reading, not compiled, not run on hardware
//
#include <hip/hip_runtime.h>
#include <hip/hip_bf16.h>

#define HIDDEN   1024
#define NHEADS   16
#define HEADDIM  64
#define BATCH    8
#define SEQ      1024
#define ROWS     (BATCH * SEQ)   // 8192

typedef __attribute__((ext_vector_type(16))) _Float16 v16h;
typedef __attribute__((ext_vector_type(8)))  _Float16 v8h;
typedef __attribute__((ext_vector_type(4)))  _Float16 v4h;
typedef __attribute__((ext_vector_type(8)))  float    v8f;

// ---------------------------------------------------------------------------
// helpers
// ---------------------------------------------------------------------------
static __device__ __forceinline__ v8f wmma_f16(v16h a, v16h b, v8f c) {
  // D = A(16x32 f16) * B(32x16 f16) + C(16x16 f32)
  return __builtin_amdgcn_wmma_f32_16x16x32_f16(
      /*neg_a=*/false, a, /*neg_b=*/false, b,
      /*c_mod=*/(short)0, c, /*reuse_a=*/false, /*reuse_b=*/false);
}

// Build a v16h fragment from two 16-byte chunks (A-layout: elems 0..7 then 8..15)
static __device__ __forceinline__ v16h ld_frag2(const _Float16* p0, const _Float16* p1) {
  v8h a = *(const v8h*)p0;
  v8h b = *(const v8h*)p1;
  v16h r;
#pragma unroll
  for (int i = 0; i < 8; ++i) { r[i] = a[i]; r[8 + i] = b[i]; }
  return r;
}

static __device__ __forceinline__ float redmax16(float x) {
  x = fmaxf(x, __shfl_xor(x, 1, 32));
  x = fmaxf(x, __shfl_xor(x, 2, 32));
  x = fmaxf(x, __shfl_xor(x, 4, 32));
  x = fmaxf(x, __shfl_xor(x, 8, 32));
  return x;
}
static __device__ __forceinline__ float redsum16(float x) {
  x += __shfl_xor(x, 1, 32);
  x += __shfl_xor(x, 2, 32);
  x += __shfl_xor(x, 4, 32);
  x += __shfl_xor(x, 8, 32);
  return x;
}

// CDNA5 async copy: global -> LDS, 32 bytes per lane (two b128 ops sharing the
// address VGPRs; INST_OFFSET applies to BOTH the LDS and global addresses).
// lds_off = low 32 bits of the flat shared-memory address (== LDS byte offset
// per the CDNA5 aperture rule for shared addresses).
static __device__ __forceinline__ void async_cp_32B(void* lds_ptr, const void* gptr) {
  unsigned lds_off = (unsigned)(uintptr_t)lds_ptr;
  unsigned long long ga = (unsigned long long)(uintptr_t)gptr;
  asm volatile(
      "global_load_async_to_lds_b128 %0, %1, off\n\t"
      "global_load_async_to_lds_b128 %0, %1, off offset:16"
      :: "v"(lds_off), "v"(ga) : "memory");
}
static __device__ __forceinline__ void wait_asynccnt0() {
  asm volatile("s_wait_asynccnt 0" ::: "memory");
}

// ---------------------------------------------------------------------------
// conversion kernels
// ---------------------------------------------------------------------------
__global__ void cvt_x_f16(const float* __restrict__ x, _Float16* __restrict__ o, int n4) {
  int i = blockIdx.x * blockDim.x + threadIdx.x;
  if (i >= n4) return;
  float4 f = ((const float4*)x)[i];
  v4h h;
  h[0] = (_Float16)f.x; h[1] = (_Float16)f.y;
  h[2] = (_Float16)f.z; h[3] = (_Float16)f.w;
  *(v4h*)(o + 4 * (size_t)i) = h;
}

// W: [K=1024][N=1024] f32 row-major -> WT: [N][K] f16
__global__ void cvt_wT_f16(const float* __restrict__ W, _Float16* __restrict__ WT) {
  int idx = blockIdx.x * blockDim.x + threadIdx.x;   // coalesced read
  int k = idx >> 10;
  int n = idx & 1023;
  WT[((size_t)n << 10) + k] = (_Float16)W[idx];
}

// ---------------------------------------------------------------------------
// GEMM:  C[M x 1024] = A[M x 1024](f16) * W(Bt = [N][K] f16) + bias(f32)
// mode 0: C f16 row-major   (Q, K, attn-merged)
// mode 1: C f16 written transposed per-head: Vt[b,h][d][S]
// mode 2: C f32 row-major   (final output)
// 128x64 block tile, BK=32; 256 threads = 8 waves (4x2); double-buffered LDS
// filled by CDNA5 async global->LDS copies (ASYNCcnt).
// ---------------------------------------------------------------------------
#define BM 128
#define BN 64
#define BK 32
#define LDSB 40   // padded row stride in halves (80B)

__global__ __launch_bounds__(256) void gemm_f16_kernel(
    const _Float16* __restrict__ A,
    const _Float16* __restrict__ Bt,
    const float* __restrict__ bias,
    void* __restrict__ C,
    int mode) {
  __shared__ __align__(16) _Float16 As[2][BM * LDSB];
  __shared__ __align__(16) _Float16 Bs[2][BN * LDSB];

  const int tid  = threadIdx.x;
  const int lane = tid & 31;
  const int wave = tid >> 5;
  const int wm   = wave >> 1;       // 0..3  (rows of 32)
  const int wn   = wave & 1;        // 0..1  (cols of 32)
  const int bm0  = blockIdx.y * BM;
  const int bn0  = blockIdx.x * BN;
  const int hi   = (lane >= 16) ? 1 : 0;
  const int l16  = lane & 15;
  const int ca   = hi ? 8 : 0;      // A-fragment k sub-offset (halves)
  const int cb   = hi ? 16 : 0;     // unused; the B fragment also uses ca

  // staging assignment: every thread covers 16 halves (32B) of the A tile,
  // threads 0..127 additionally cover the B tile.
  const int arow = tid >> 1;
  const int aseg = (tid & 1) * 16;
  const int brow = (tid & 127) >> 1;
  const int bseg = ((tid & 127) & 1) * 16;
  const bool doB = (tid < 128);

  v8f acc[2][2];
#pragma unroll
  for (int i = 0; i < 2; ++i)
#pragma unroll
    for (int j = 0; j < 2; ++j)
#pragma unroll
      for (int e = 0; e < 8; ++e) acc[i][j][e] = 0.0f;

  // ---- prologue: stage k-chunk 0 into buffer 0
  async_cp_32B(&As[0][arow * LDSB + aseg],
               A + ((size_t)(bm0 + arow) << 10) + aseg);
  if (doB)
    async_cp_32B(&Bs[0][brow * LDSB + bseg],
                 Bt + ((size_t)(bn0 + brow) << 10) + bseg);

  int cur = 0;
  for (int k0 = 0; k0 < HIDDEN; k0 += BK) {
    wait_asynccnt0();     // our async copies into buf[cur] have landed
    __syncthreads();      // everyone's copies have landed

    // ---- stage next k-chunk into the other buffer while we compute
    if (k0 + BK < HIDDEN) {
      async_cp_32B(&As[cur ^ 1][arow * LDSB + aseg],
                   A + ((size_t)(bm0 + arow) << 10) + (k0 + BK) + aseg);
      if (doB)
        async_cp_32B(&Bs[cur ^ 1][brow * LDSB + bseg],
                     Bt + ((size_t)(bn0 + brow) << 10) + (k0 + BK) + bseg);
    }

    // ---- 4 WMMAs from buf[cur]
#pragma unroll
    for (int mi = 0; mi < 2; ++mi) {
      int r = wm * 32 + mi * 16 + l16;
      v16h af = ld_frag2(&As[cur][r * LDSB + ca], &As[cur][r * LDSB + ca + 16]);
#pragma unroll
      for (int ni = 0; ni < 2; ++ni) {
        int c = wn * 32 + ni * 16 + l16;
        v16h bf = ld_frag2(&Bs[cur][c * LDSB + ca], &Bs[cur][c * LDSB + ca + 16]);
        acc[mi][ni] = wmma_f16(af, bf, acc[mi][ni]);
      }
    }
    __syncthreads();      // all waves done reading buf[cur] before it's refilled
    cur ^= 1;
  }

  // ---- epilogue: bias + store
#pragma unroll
  for (int mi = 0; mi < 2; ++mi) {
#pragma unroll
    for (int ni = 0; ni < 2; ++ni) {
      int col  = bn0 + wn * 32 + ni * 16 + l16;
      float bc = bias[col];
      int rb   = bm0 + wm * 32 + mi * 16 + (hi ? 8 : 0);
#pragma unroll
      for (int j = 0; j < 8; ++j) {
        float v = acc[mi][ni][j] + bc;
        int row = rb + j;
        if (mode == 2) {
          ((float*)C)[((size_t)row << 10) + col] = v;
        } else if (mode == 0) {
          ((_Float16*)C)[((size_t)row << 10) + col] = (_Float16)v;
        } else {
          // V transposed per head: Vt[(b*16+h)*64 + d][s]
          int b = row >> 10, s = row & 1023;
          int h = col >> 6,  d = col & 63;
          ((_Float16*)C)[(((size_t)(b * NHEADS + h) * HEADDIM + d) << 10) + s] =
              (_Float16)v;
        }
      }
    }
  }
}

// ---------------------------------------------------------------------------
// Flash attention: one wave per 16-query tile of one (b,h).
// Q,K: [B*S][1024] f16 ; Vt: [(b*16+h)*64 + d][S] f16 ; O: [B*S][1024] f16
// ---------------------------------------------------------------------------
__global__ __launch_bounds__(256) void attn_kernel(
    const _Float16* __restrict__ Q,
    const _Float16* __restrict__ K,
    const _Float16* __restrict__ Vt,
    _Float16* __restrict__ O) {
  __shared__ __align__(32) _Float16 pbuf[8][16 * LDSB];  // per-wave P transpose buf

  const int lane = threadIdx.x & 31;
  const int wave = threadIdx.x >> 5;
  const int hi   = (lane >= 16) ? 1 : 0;
  const int l16  = lane & 15;
  const int ca   = hi ? 8 : 0;
  const int bh   = blockIdx.y;           // b*16 + h
  const int b    = bh >> 4;
  const int h    = bh & 15;
  const int q0   = (blockIdx.x * 8 + wave) * 16;

  // --- Q fragments (A-matrix 16x32 layout), K-dim = head dim 64 -> 2 frags
  const _Float16* qrow = Q + (((size_t)b * SEQ + q0 + l16) << 10) + h * HEADDIM;
  v16h qf0 = ld_frag2(qrow + 2 * ca,      qrow + 2 * ca + 8);
  v16h qf1 = ld_frag2(qrow + 2 * ca + 32, qrow + 2 * ca + 40);

  float m[8], l[8];
  v8f o[4];
#pragma unroll
  for (int j = 0; j < 8; ++j) { m[j] = -1e30f; l[j] = 0.0f; }
#pragma unroll
  for (int ni = 0; ni < 4; ++ni)
#pragma unroll
    for (int e = 0; e < 8; ++e) o[ni][e] = 0.0f;

  _Float16* pb = pbuf[wave];
  const float scale = 0.125f;            // 1/sqrt(64)
  const _Float16* kbase = K + (((size_t)b * SEQ) << 10) + h * HEADDIM;
  const _Float16* vbase = Vt + (((size_t)bh * HEADDIM) << 10);

  for (int kb = 0; kb < SEQ; kb += 32) {
    // warm the caches for the next chunk (global_prefetch)
    if (kb + 32 < SEQ) {
      __builtin_prefetch(kbase + (((size_t)(kb + 32 + l16)) << 10), 0, 0);
      __builtin_prefetch(vbase + (((size_t)l16) << 10) + kb + 32, 0, 0);
    }

    // ---- scores S = Q K^T : B-matrix of WMMA is K^T (d x keys), whose
    //      N-major layout is exactly K stored row-major -> direct v16h loads.
    const _Float16* kr0 = kbase + (((size_t)(kb + l16)) << 10) + (hi ? 16 : 0);
    const _Float16* kr1 = kbase + (((size_t)(kb + 16 + l16)) << 10) + (hi ? 16 : 0);
    v16h kf00 = *(const v16h*)(kr0);        // keys kb..kb+15,  d 0..31
    v16h kf01 = *(const v16h*)(kr0 + 32);   // keys kb..kb+15,  d 32..63
    v16h kf10 = *(const v16h*)(kr1);        // keys kb+16..31,  d 0..31
    v16h kf11 = *(const v16h*)(kr1 + 32);

    v8f s0, s1;
#pragma unroll
    for (int e = 0; e < 8; ++e) { s0[e] = 0.0f; s1[e] = 0.0f; }
    s0 = wmma_f16(qf0, kf00, s0);
    s0 = wmma_f16(qf1, kf01, s0);
    s1 = wmma_f16(qf0, kf10, s1);
    s1 = wmma_f16(qf1, kf11, s1);

    // ---- online softmax over this 32-key chunk.
    //      C-layout: VGPR j holds row M=j+8*hi, key N=l16 (s0) / 16+l16 (s1)
#pragma unroll
    for (int j = 0; j < 8; ++j) {
      float x0 = s0[j] * scale;
      float x1 = s1[j] * scale;
      float mj = redmax16(fmaxf(x0, x1));
      float mn = fmaxf(m[j], mj);
      float rs = __expf(m[j] - mn);
      m[j] = mn;
      float p0 = __expf(x0 - mn);
      float p1 = __expf(x1 - mn);
      l[j] = l[j] * rs + redsum16(p0 + p1);
#pragma unroll
      for (int ni = 0; ni < 4; ++ni) o[ni][j] *= rs;
      int prow = j + (hi ? 8 : 0);
      pb[prow * LDSB + l16]      = (_Float16)p0;
      pb[prow * LDSB + 16 + l16] = (_Float16)p1;
    }

    // all LDS stores from this wave must land before the transpose reads
    asm volatile("s_wait_dscnt 0" ::: "memory");

    // ---- P (16x32) as A-fragment, then O += P * V  (4 d-chunks of 16)
    v16h pf = ld_frag2(&pb[l16 * LDSB + 2 * ca], &pb[l16 * LDSB + 2 * ca + 8]);
#pragma unroll
    for (int ni = 0; ni < 4; ++ni) {
      const _Float16* vp =
          vbase + (((size_t)(ni * 16 + l16)) << 10) + kb + (hi ? 16 : 0);
      v16h vf = *(const v16h*)vp;           // Vt rows: contiguous keys
      o[ni] = wmma_f16(pf, vf, o[ni]);
    }
  }

  // ---- finalize: divide by row sum, write merged [B,S,H] f16
#pragma unroll
  for (int j = 0; j < 8; ++j) {
    float inv = 1.0f / l[j];
    int row = q0 + j + (hi ? 8 : 0);
#pragma unroll
    for (int ni = 0; ni < 4; ++ni) {
      O[(((size_t)b * SEQ + row) << 10) + h * HEADDIM + ni * 16 + l16] =
          (_Float16)(o[ni][j] * inv);
    }
  }
}

// ---------------------------------------------------------------------------
// host launcher
// ---------------------------------------------------------------------------
extern "C" void kernel_launch(void* const* d_in, const int* in_sizes, int n_in,
                              void* d_out, int out_size, void* d_ws, size_t ws_size,
                              hipStream_t stream) {
  const float* x  = (const float*)d_in[0];
  const float* Wq = (const float*)d_in[1];
  const float* bq = (const float*)d_in[2];
  const float* Wk = (const float*)d_in[3];
  const float* bk = (const float*)d_in[4];
  const float* Wv = (const float*)d_in[5];
  const float* bv = (const float*)d_in[6];
  const float* Wo = (const float*)d_in[7];
  const float* bo = (const float*)d_in[8];

  char* ws = (char*)d_ws;
  const size_t SZ_X = (size_t)ROWS * HIDDEN * sizeof(_Float16);   // 16 MB
  const size_t SZ_W = (size_t)HIDDEN * HIDDEN * sizeof(_Float16); //  2 MB
  _Float16* Xh  = (_Float16*)(ws);
  _Float16* WqT = (_Float16*)(ws + SZ_X);
  _Float16* WkT = (_Float16*)(ws + SZ_X + SZ_W);
  _Float16* WvT = (_Float16*)(ws + SZ_X + 2 * SZ_W);
  _Float16* WoT = (_Float16*)(ws + SZ_X + 3 * SZ_W);
  _Float16* Qh  = (_Float16*)(ws + SZ_X + 4 * SZ_W);
  _Float16* Kh  = (_Float16*)(ws + 2 * SZ_X + 4 * SZ_W);
  _Float16* Vt  = (_Float16*)(ws + 3 * SZ_X + 4 * SZ_W);
  _Float16* Ah  = (_Float16*)(ws + 4 * SZ_X + 4 * SZ_W);

  // 1) convert activations + weights to f16 (weights transposed to N-major)
  cvt_x_f16<<<(ROWS * HIDDEN / 4 + 255) / 256, 256, 0, stream>>>(x, Xh,
                                                                 ROWS * HIDDEN / 4);
  cvt_wT_f16<<<HIDDEN * HIDDEN / 256, 256, 0, stream>>>(Wq, WqT);
  cvt_wT_f16<<<HIDDEN * HIDDEN / 256, 256, 0, stream>>>(Wk, WkT);
  cvt_wT_f16<<<HIDDEN * HIDDEN / 256, 256, 0, stream>>>(Wv, WvT);
  cvt_wT_f16<<<HIDDEN * HIDDEN / 256, 256, 0, stream>>>(Wo, WoT);

  // 2) QKV projections (WMMA GEMMs, async-LDS staged)
  dim3 ggrid(HIDDEN / BN, ROWS / BM);   // (16, 64)
  gemm_f16_kernel<<<ggrid, 256, 0, stream>>>(Xh, WqT, bq, (void*)Qh, 0);
  gemm_f16_kernel<<<ggrid, 256, 0, stream>>>(Xh, WkT, bk, (void*)Kh, 0);
  gemm_f16_kernel<<<ggrid, 256, 0, stream>>>(Xh, WvT, bv, (void*)Vt, 1);

  // 3) flash attention
  attn_kernel<<<dim3(SEQ / (16 * 8), BATCH * NHEADS), 256, 0, stream>>>(Qh, Kh, Vt, Ah);

  // 4) output projection -> f32
  gemm_f16_kernel<<<ggrid, 256, 0, stream>>>(Ah, WoT, bo, d_out, 2);
}
